// HeteroGATLayerReal_52166672777270
// MI455X (gfx1250) — compile-verified, matrix operand order checked by reading, not run on hardware
//
#include <hip/hip_runtime.h>
#include <math.h>

#define NP 50000
#define NA 25000
#define DD 64
#define ALPHA 0.2f

typedef __attribute__((ext_vector_type(16))) _Float16 v16h;
typedef __attribute__((ext_vector_type(8)))  _Float16 v8h;
typedef __attribute__((ext_vector_type(8)))  float    v8f;

// ---------------------------------------------------------------------------
// Convert f32 -> f16 (row-major features)
// ---------------------------------------------------------------------------
__global__ void cvt_f16(const float* __restrict__ x, _Float16* __restrict__ y, size_t n)
{
    const size_t i = (size_t)blockIdx.x * blockDim.x + threadIdx.x;
    if (i < n) y[i] = (_Float16)x[i];
}

// ---------------------------------------------------------------------------
// Pack one 64x64 weight into WMMA B-fragment order (f16):
//   slot s = ((wid*2 + c)*32 + lane)*16 + h
//   N = wid*16 + (lane&15);  kb = (lane>>4)*8
//   K = c*32 + (h<8 ? kb+h : 16+kb+(h-8))
// so the GEMM reads one aligned 32B v16h per (wave, chunk).
// ---------------------------------------------------------------------------
__global__ void pack_w(const float* __restrict__ W, _Float16* __restrict__ Bp)
{
    const int s = blockIdx.x * blockDim.x + threadIdx.x;
    if (s >= 4096) return;
    const int wid  = s >> 10;
    const int c    = (s >> 9) & 1;
    const int lane = (s >> 4) & 31;
    const int h    = s & 15;
    const int kb   = (lane >> 4) * 8;
    const int K    = c * 32 + ((h < 8) ? (kb + h) : (16 + kb + (h - 8)));
    const int N    = wid * 16 + (lane & 15);
    Bp[s] = (_Float16)W[K * DD + N];
}

// ---------------------------------------------------------------------------
// Fused 4-output GEMM: Yj[M,64] = Xh[M,64] @ Wj + bj, j=0..3
// One wave = one 16x16 tile of all four outputs; A fragment loaded once,
// 4 WMMAs per K=32 chunk -> 8 v_wmma_f32_16x16x32_f16 per wave.
// ---------------------------------------------------------------------------
__global__ void gemm64x4_wmma(const _Float16* __restrict__ Xh,
                              const _Float16* __restrict__ Bp,   // 4 * 4096 halves
                              const float* __restrict__ b0, const float* __restrict__ b1,
                              const float* __restrict__ b2, const float* __restrict__ b3,
                              float* __restrict__ Y0, float* __restrict__ Y1,
                              float* __restrict__ Y2, float* __restrict__ Y3, int M)
{
    const int lane  = threadIdx.x & 31;
    const int wid   = threadIdx.x >> 5;      // N tile 0..3
    const int laneM = lane & 15;
    const int hi    = lane >> 4;
    const int kb    = hi * 8;
    int row = blockIdx.x * 16 + laneM;
    if (row >= M) row = M - 1;               // clamp; rows >= M never stored
    const _Float16* xr = Xh + (size_t)row * DD;
    const int col = wid * 16 + laneM;

    v8f acc0 = {0.f,0.f,0.f,0.f,0.f,0.f,0.f,0.f};
    v8f acc1 = acc0, acc2 = acc0, acc3 = acc0;

#pragma unroll
    for (int c = 0; c < 2; ++c) {
        const int k0 = c * 32;
        const v8h alo = *(const v8h*)(xr + k0 + kb);        // K = k0+kb .. +7
        const v8h ahi = *(const v8h*)(xr + k0 + 16 + kb);   // K = k0+16+kb .. +7
        const v16h a = __builtin_shufflevector(alo, ahi,
                          0,1,2,3,4,5,6,7,8,9,10,11,12,13,14,15);
        const size_t boff = ((size_t)(wid * 2 + c) * 32 + lane) * 16;
        const v16h w0 = *(const v16h*)(Bp          + boff);
        const v16h w1 = *(const v16h*)(Bp + 4096   + boff);
        const v16h w2 = *(const v16h*)(Bp + 2*4096 + boff);
        const v16h w3 = *(const v16h*)(Bp + 3*4096 + boff);
        acc0 = __builtin_amdgcn_wmma_f32_16x16x32_f16(false, a, false, w0, (short)0, acc0, false, false);
        acc1 = __builtin_amdgcn_wmma_f32_16x16x32_f16(false, a, false, w1, (short)0, acc1, false, false);
        acc2 = __builtin_amdgcn_wmma_f32_16x16x32_f16(false, a, false, w2, (short)0, acc2, false, false);
        acc3 = __builtin_amdgcn_wmma_f32_16x16x32_f16(false, a, false, w3, (short)0, acc3, false, false);
    }

    const float bb0 = b0[col], bb1 = b1[col], bb2 = b2[col], bb3 = b3[col];
#pragma unroll
    for (int r = 0; r < 8; ++r) {
        const int m = blockIdx.x * 16 + r + 8 * hi;
        if (m < M) {
            const size_t o = (size_t)m * DD + col;
            Y0[o] = acc0[r] + bb0;
            Y1[o] = acc1[r] + bb1;
            Y2[o] = acc2[r] + bb2;
            Y3[o] = acc3[r] + bb3;
        }
    }
}

// ---------------------------------------------------------------------------
// State self-transform: Y[1,64] = feat_S @ W + b  (tiny; 64 threads)
// ---------------------------------------------------------------------------
__global__ void state_gemm(const float* __restrict__ fS, const float* __restrict__ W,
                           const float* __restrict__ b, float* __restrict__ Y)
{
    const int n = threadIdx.x;
    if (n >= DD) return;
    float s = b[n];
#pragma unroll
    for (int k = 0; k < DD; ++k) s += fS[k] * W[k * DD + n];
    Y[n] = s;
}

// ---------------------------------------------------------------------------
// Per-node attention scalar: out[n] = dot(Z[n,:], a[0:64]).  Wave per row.
// ---------------------------------------------------------------------------
__global__ void rowdot64(const float* __restrict__ Z, const float* __restrict__ a,
                         float* __restrict__ out, int n)
{
    const int w    = blockIdx.x * (blockDim.x >> 5) + (threadIdx.x >> 5);
    const int lane = threadIdx.x & 31;
    if (w >= n) return;
    const float2 z2 = ((const float2*)(Z + (size_t)w * DD))[lane];
    const float2 a2 = ((const float2*)a)[lane];
    float s = z2.x * a2.x + z2.y * a2.y;
#pragma unroll
    for (int off = 16; off > 0; off >>= 1) s += __shfl_down(s, off, 32);
    if (lane == 0) out[w] = s;
}

// ---------------------------------------------------------------------------
// Workspace init
// ---------------------------------------------------------------------------
__global__ void fill_i32(int* __restrict__ p, int v, int n)
{
    const int i = blockIdx.x * blockDim.x + threadIdx.x;
    if (i < n) p[i] = v;
}
__global__ void fill_f32(float* __restrict__ p, float v, size_t n)
{
    const size_t i = (size_t)blockIdx.x * blockDim.x + threadIdx.x;
    if (i < n) p[i] = v;
}

// Monotonic int encoding of float for atomicMax-based segment max.
__device__ __forceinline__ int f2key(float f)
{
    const int i = __float_as_int(f);
    return (i >= 0) ? i : (i ^ 0x7fffffff);
}
__device__ __forceinline__ float key2f(int k)
{
    const int i = (k >= 0) ? k : (k ^ 0x7fffffff);
    return __int_as_float(i);
}

// ---------------------------------------------------------------------------
// Edge pass 1: score + leaky_relu + segment max (encoded-int atomicMax)
// ---------------------------------------------------------------------------
__global__ void edge_max(const int* __restrict__ src, const int* __restrict__ dst,
                         const float* __restrict__ esrc, const float* __restrict__ edst,
                         float* __restrict__ e_buf, int* __restrict__ mkey, int E)
{
    const int i = blockIdx.x * blockDim.x + threadIdx.x;
    if (i >= E) return;
    float e = esrc[src[i]] + edst[dst[i]];
    e = (e > 0.f) ? e : ALPHA * e;
    e_buf[i] = e;
    atomicMax(&mkey[dst[i]], f2key(e));
}

// ---------------------------------------------------------------------------
// Edge pass 2: ex = exp(e - m[dst]); den[dst] += ex
// ---------------------------------------------------------------------------
__global__ void edge_expsum(const int* __restrict__ dst, const int* __restrict__ mkey,
                            float* __restrict__ e_buf, float* __restrict__ den, int E)
{
    const int i = blockIdx.x * blockDim.x + threadIdx.x;
    if (i >= E) return;
    const int d = dst[i];
    const float ex = __expf(e_buf[i] - key2f(mkey[d]));
    e_buf[i] = ex;
    atomicAdd(&den[d], ex);
}

// ---------------------------------------------------------------------------
// Edge pass 3: acc[dst,:] += (ex/den[dst]) * Z[src,:]   (wave per edge,
// float2 per lane -> one coalesced 256B row gather + 64 f32 atomics)
// ---------------------------------------------------------------------------
__global__ void edge_scatter(const int* __restrict__ src, const int* __restrict__ dst,
                             const float* __restrict__ e_buf, const float* __restrict__ den,
                             const float* __restrict__ Z, float* __restrict__ acc, int E)
{
    const int w    = blockIdx.x * (blockDim.x >> 5) + (threadIdx.x >> 5);
    const int lane = threadIdx.x & 31;
    if (w >= E) return;
    const int s = src[w], d = dst[w];
    const float wgt = e_buf[w] / den[d];
    const float2 z2 = ((const float2*)(Z + (size_t)s * DD))[lane];
    float* ar = acc + (size_t)d * DD + 2 * lane;
    atomicAdd(ar,     wgt * z2.x);
    atomicAdd(ar + 1, wgt * z2.y);
}

// ---------------------------------------------------------------------------
// h = relu(wh + a1 + a2)
// ---------------------------------------------------------------------------
__global__ void finalize_relu(const float* __restrict__ wh,
                              const float* __restrict__ a1,
                              const float* __restrict__ a2,
                              float* __restrict__ out, size_t n)
{
    const size_t i = (size_t)blockIdx.x * blockDim.x + threadIdx.x;
    if (i >= n) return;
    const float v = wh[i] + a1[i] + a2[i];
    out[i] = (v > 0.f) ? v : 0.f;
}

extern "C" void kernel_launch(void* const* d_in, const int* in_sizes, int n_in,
                              void* d_out, int out_size, void* d_ws, size_t ws_size,
                              hipStream_t stream)
{
    (void)n_in; (void)out_size; (void)ws_size;

    const float* feat_P = (const float*)d_in[0];
    const float* feat_A = (const float*)d_in[1];
    const float* feat_S = (const float*)d_in[2];
    const float* W_P   = (const float*)d_in[3];  const float* b_P   = (const float*)d_in[4];
    const float* W_A   = (const float*)d_in[5];  const float* b_A   = (const float*)d_in[6];
    const float* W_p2p = (const float*)d_in[7];  const float* b_p2p = (const float*)d_in[8];
    const float* W_p2a = (const float*)d_in[9];  const float* b_p2a = (const float*)d_in[10];
    const float* W_a2p = (const float*)d_in[11]; const float* b_a2p = (const float*)d_in[12];
    const float* W_a2a = (const float*)d_in[13]; const float* b_a2a = (const float*)d_in[14];
    const float* W_p2s = (const float*)d_in[15]; const float* b_p2s = (const float*)d_in[16];
    const float* W_a2s = (const float*)d_in[17]; const float* b_a2s = (const float*)d_in[18];
    const float* W_in  = (const float*)d_in[19]; const float* b_in  = (const float*)d_in[20];
    const float* a_p2p = (const float*)d_in[21];
    const float* a_p2a = (const float*)d_in[22];
    const float* a_a2p = (const float*)d_in[23];
    const float* a_a2a = (const float*)d_in[24];
    const float* a_p2s = (const float*)d_in[25];
    const float* a_a2s = (const float*)d_in[26];
    const int* src_p2p = (const int*)d_in[27]; const int* dst_p2p = (const int*)d_in[28];
    const int* src_p2a = (const int*)d_in[29]; const int* dst_p2a = (const int*)d_in[30];
    const int* src_a2p = (const int*)d_in[31]; const int* dst_a2p = (const int*)d_in[32];
    const int* src_a2a = (const int*)d_in[33]; const int* dst_a2a = (const int*)d_in[34];
    const int* src_p2s = (const int*)d_in[35]; const int* dst_p2s = (const int*)d_in[36];
    const int* src_a2s = (const int*)d_in[37]; const int* dst_a2s = (const int*)d_in[38];

    const int E_p2p = in_sizes[27], E_p2a = in_sizes[29];
    const int E_a2p = in_sizes[31], E_a2a = in_sizes[33];
    const int E_p2s = in_sizes[35], E_a2s = in_sizes[37];

    float* out = (float*)d_out;

    // ---- workspace carve-up (float units, 32B-aligned blocks) ----
    float* ws = (float*)d_ws;
    size_t off = 0;
    auto alloc = [&](size_t nfloats) {
        off = (off + 7) & ~(size_t)7;         // 32-byte alignment
        float* p = ws + off; off += nfloats; return p;
    };

    float* Wh_P  = alloc((size_t)NP * DD);
    float* Wh_A  = alloc((size_t)NA * DD);
    float* Wh_S  = alloc(DD);
    float* z_p2p = alloc((size_t)NP * DD);
    float* z_p2a = alloc((size_t)NP * DD);
    float* z_p2s = alloc((size_t)NP * DD);
    float* z_a2p = alloc((size_t)NA * DD);
    float* z_a2a = alloc((size_t)NA * DD);
    float* z_a2s = alloc((size_t)NA * DD);

    _Float16* XhP = (_Float16*)alloc((size_t)NP * DD / 2);
    _Float16* XhA = (_Float16*)alloc((size_t)NA * DD / 2);
    _Float16* BpP = (_Float16*)alloc(4 * 4096 / 2);    // W_P, W_p2p, W_p2a, W_p2s
    _Float16* BpA = (_Float16*)alloc(4 * 4096 / 2);    // W_A, W_a2p, W_a2a, W_a2s

    float* es_p2p = alloc(NP); float* es_p2a = alloc(NP); float* es_p2s = alloc(NP);
    float* es_a2p = alloc(NA); float* es_a2a = alloc(NA); float* es_a2s = alloc(NA);
    float* ed_p2p = alloc(NP); float* ed_a2p = alloc(NP);
    float* ed_p2a = alloc(NA); float* ed_a2a = alloc(NA);
    float* ed_p2s = alloc(1);  float* ed_a2s = alloc(1);

    const int NMK = NP + NA + NP + NA + 1 + 1;
    int* mk_base = (int*)alloc(NMK);
    int* mk_p2p = mk_base;            int* mk_p2a = mk_p2p + NP;
    int* mk_a2p = mk_p2a + NA;        int* mk_a2a = mk_a2p + NP;
    int* mk_p2s = mk_a2a + NA;        int* mk_a2s = mk_p2s + 1;
    float* dn_base = alloc(NMK);
    float* dn_p2p = dn_base;          float* dn_p2a = dn_p2p + NP;
    float* dn_a2p = dn_p2a + NA;      float* dn_a2a = dn_a2p + NP;
    float* dn_p2s = dn_a2a + NA;      float* dn_a2s = dn_p2s + 1;

    float* ex_p2p = alloc(E_p2p); float* ex_p2a = alloc(E_p2a);
    float* ex_a2p = alloc(E_a2p); float* ex_a2a = alloc(E_a2a);
    float* ex_p2s = alloc(E_p2s); float* ex_a2s = alloc(E_a2s);

    const size_t NACC = (size_t)(2 * NP + 2 * NA) * DD + 2 * DD;
    float* acc_base = alloc(NACC);
    float* ac_p2p = acc_base;                 float* ac_a2p = ac_p2p + (size_t)NP * DD;
    float* ac_p2a = ac_a2p + (size_t)NP * DD; float* ac_a2a = ac_p2a + (size_t)NA * DD;
    float* ac_p2s = ac_a2a + (size_t)NA * DD; float* ac_a2s = ac_p2s + DD;

    const int TB = 256;
    auto cdiv = [](long long a, long long b) { return (int)((a + b - 1) / b); };

    // ---- 0) precision staging: f16 features + packed B fragments ----
    cvt_f16<<<cdiv((long long)NP * DD, TB), TB, 0, stream>>>(feat_P, XhP, (size_t)NP * DD);
    cvt_f16<<<cdiv((long long)NA * DD, TB), TB, 0, stream>>>(feat_A, XhA, (size_t)NA * DD);
    pack_w<<<16, TB, 0, stream>>>(W_P,   BpP);
    pack_w<<<16, TB, 0, stream>>>(W_p2p, BpP + 4096);
    pack_w<<<16, TB, 0, stream>>>(W_p2a, BpP + 2 * 4096);
    pack_w<<<16, TB, 0, stream>>>(W_p2s, BpP + 3 * 4096);
    pack_w<<<16, TB, 0, stream>>>(W_A,   BpA);
    pack_w<<<16, TB, 0, stream>>>(W_a2p, BpA + 4096);
    pack_w<<<16, TB, 0, stream>>>(W_a2a, BpA + 2 * 4096);
    pack_w<<<16, TB, 0, stream>>>(W_a2s, BpA + 3 * 4096);

    // ---- 1) node transforms: fused 4-output WMMA GEMMs ----
    gemm64x4_wmma<<<cdiv(NP, 16), 128, 0, stream>>>(
        XhP, BpP, b_P, b_p2p, b_p2a, b_p2s, Wh_P, z_p2p, z_p2a, z_p2s, NP);
    gemm64x4_wmma<<<cdiv(NA, 16), 128, 0, stream>>>(
        XhA, BpA, b_A, b_a2p, b_a2a, b_a2s, Wh_A, z_a2p, z_a2a, z_a2s, NA);
    state_gemm<<<1, 64, 0, stream>>>(feat_S, W_in, b_in, Wh_S);

    // ---- 2) per-node attention scalars ----
    auto dot = [&](const float* Z, const float* a, float* o, int n) {
        rowdot64<<<cdiv(n, 8), TB, 0, stream>>>(Z, a, o, n);
    };
    dot(z_p2p, a_p2p,      es_p2p, NP);  dot(Wh_P, a_p2p + DD, ed_p2p, NP);
    dot(z_p2a, a_p2a,      es_p2a, NP);  dot(Wh_A, a_p2a + DD, ed_p2a, NA);
    dot(z_a2p, a_a2p,      es_a2p, NA);  dot(Wh_P, a_a2p + DD, ed_a2p, NP);
    dot(z_a2a, a_a2a,      es_a2a, NA);  dot(Wh_A, a_a2a + DD, ed_a2a, NA);
    dot(z_p2s, a_p2s,      es_p2s, NP);  dot(Wh_S, a_p2s + DD, ed_p2s, 1);
    dot(z_a2s, a_a2s,      es_a2s, NA);  dot(Wh_S, a_a2s + DD, ed_a2s, 1);

    // ---- 3) init reduction buffers ----
    fill_i32<<<cdiv(NMK, TB), TB, 0, stream>>>(mk_base, (int)0x80000000, NMK);
    fill_f32<<<cdiv(NMK, TB), TB, 0, stream>>>(dn_base, 0.f, (size_t)NMK);
    fill_f32<<<cdiv((long long)NACC, TB), TB, 0, stream>>>(acc_base, 0.f, NACC);

    // ---- 4) edge passes: max -> exp/sum -> weighted scatter ----
    auto rel = [&](const int* src, const int* dst, int E,
                   const float* es, const float* ed,
                   int* mk, float* dn, float* ex,
                   const float* Z, float* ac) {
        edge_max    <<<cdiv(E, TB), TB, 0, stream>>>(src, dst, es, ed, ex, mk, E);
        edge_expsum <<<cdiv(E, TB), TB, 0, stream>>>(dst, mk, ex, dn, E);
        edge_scatter<<<cdiv(E, 8),  TB, 0, stream>>>(src, dst, ex, dn, Z, ac, E);
    };
    rel(src_p2p, dst_p2p, E_p2p, es_p2p, ed_p2p, mk_p2p, dn_p2p, ex_p2p, z_p2p, ac_p2p);
    rel(src_p2a, dst_p2a, E_p2a, es_p2a, ed_p2a, mk_p2a, dn_p2a, ex_p2a, z_p2a, ac_p2a);
    rel(src_a2p, dst_a2p, E_a2p, es_a2p, ed_a2p, mk_a2p, dn_a2p, ex_a2p, z_a2p, ac_a2p);
    rel(src_a2a, dst_a2a, E_a2a, es_a2a, ed_a2a, mk_a2a, dn_a2a, ex_a2a, z_a2a, ac_a2a);
    rel(src_p2s, dst_p2s, E_p2s, es_p2s, ed_p2s, mk_p2s, dn_p2s, ex_p2s, z_p2s, ac_p2s);
    rel(src_a2s, dst_a2s, E_a2s, es_a2s, ed_a2s, mk_a2s, dn_a2s, ex_a2s, z_a2s, ac_a2s);

    // ---- 5) combine + relu into packed output ----
    const size_t nP = (size_t)NP * DD, nA = (size_t)NA * DD;
    finalize_relu<<<cdiv((long long)nP, TB), TB, 0, stream>>>(Wh_P, ac_p2p, ac_a2p, out, nP);
    finalize_relu<<<cdiv((long long)nA, TB), TB, 0, stream>>>(Wh_A, ac_p2a, ac_a2a, out + nP, nA);
    finalize_relu<<<1, TB, 0, stream>>>(Wh_S, ac_p2s, ac_a2s, out + nP + nA, (size_t)DD);
}
